// KnowledgeReasoningModule_37864431681723
// MI455X (gfx1250) — matrix operand order repaired and checked by reading, not compiled, not run on hardware
//
#include <hip/hip_runtime.h>
#include <hip/hip_bf16.h>
#include <cstddef>

// ---------------------------------------------------------------------------
// MI455X (gfx1250) implementation.
// All heavy GEMMs / attention contractions go through v_wmma_f32_16x16x32_bf16
// (fp32 accumulate). Softmax / LayerNorm / GELU stay in fp32 VALU.
// ---------------------------------------------------------------------------

typedef __attribute__((ext_vector_type(16))) __bf16 v16bf;
typedef __attribute__((ext_vector_type(8)))  float  v8f;

union Frag16 {
    v16bf v;
    uint4 q[2];
    __bf16 e[16];
};

#define WMMA_BF16(Afrag, Bfrag, Cacc) \
    __builtin_amdgcn_wmma_f32_16x16x32_bf16(false, (Afrag), false, (Bfrag), (short)0, (Cacc), false, false)

__device__ __forceinline__ float gelu_exact(float x) {
    return 0.5f * x * (1.0f + erff(x * 0.70710678118654752440f));
}

// ---------------------------------------------------------------------------
// Generic GEMM: C[M,N] = epi(A[M,K] @ B + bias + resid)
// TRANSB=false: B is [K,N] row-major. TRANSB=true: B is [N,K] row-major.
// Block tile 128x64, BK=32, 256 threads (8 waves, each 32x32 output).
// Double-buffered LDS + register prefetch: one barrier per k-step, global
// b128 loads of the next tile overlap the current tile's WMMAs.
// All call sites satisfy M%128==0, N%64==0, K%32==0.
// ---------------------------------------------------------------------------
template <bool TRANSB, bool GELU>
__global__ __launch_bounds__(256) void gemm_kernel(
    const float* __restrict__ A, const float* __restrict__ B,
    const float* __restrict__ bias, const float* __restrict__ resid,
    float* __restrict__ C, int M, int N, int K)
{
    constexpr int BM = 128, BN = 64, BK = 32;
    constexpr int LDA = 40, LDB = 40;     // padded rows (elements)
    __shared__ __bf16 As[2][BM * LDA];
    __shared__ __bf16 Bs[2][BN * LDB];    // stored transposed: [n][k]

    const int tid  = threadIdx.x;
    const int w    = tid >> 5;
    const int lane = tid & 31;
    const int hl   = lane >> 4;          // half-wave: 0 or 1
    const int l16  = lane & 15;
    const int wm   = w >> 1;             // 0..3  (row group of 32)
    const int wn   = w & 1;              // 0..1  (col group of 32)
    const int bm   = blockIdx.y * BM;
    const int bn   = blockIdx.x * BN;

    auto load_tiles = [&](float (&ta)[16], float (&tb)[8], int k0) {
        #pragma unroll
        for (int j = 0; j < 4; ++j) {            // A: 128x32 = 1024 float4
            int v = tid + j * 256;
            int r = v >> 3, c4 = v & 7;
            float4 f = *reinterpret_cast<const float4*>(
                &A[(size_t)(bm + r) * K + k0 + c4 * 4]);
            ta[j*4+0] = f.x; ta[j*4+1] = f.y; ta[j*4+2] = f.z; ta[j*4+3] = f.w;
        }
        #pragma unroll
        for (int j = 0; j < 2; ++j) {            // B: 32x64 = 512 float4
            int v = tid + j * 256;
            float4 f;
            if (TRANSB) {
                int n = v >> 3, c4 = v & 7;
                f = *reinterpret_cast<const float4*>(
                    &B[(size_t)(bn + n) * K + k0 + c4 * 4]);
            } else {
                int kk = v >> 4, c4 = v & 15;
                f = *reinterpret_cast<const float4*>(
                    &B[(size_t)(k0 + kk) * N + bn + c4 * 4]);
            }
            tb[j*4+0] = f.x; tb[j*4+1] = f.y; tb[j*4+2] = f.z; tb[j*4+3] = f.w;
        }
    };

    auto stage = [&](int buf, const float (&ta)[16], const float (&tb)[8]) {
        #pragma unroll
        for (int j = 0; j < 4; ++j) {
            int v = tid + j * 256;
            int r = v >> 3, c4 = v & 7;
            union { __bf16 h[4]; uint2 u; } t;
            t.h[0] = (__bf16)ta[j*4+0]; t.h[1] = (__bf16)ta[j*4+1];
            t.h[2] = (__bf16)ta[j*4+2]; t.h[3] = (__bf16)ta[j*4+3];
            *reinterpret_cast<uint2*>(&As[buf][r * LDA + c4 * 4]) = t.u;
        }
        #pragma unroll
        for (int j = 0; j < 2; ++j) {
            int v = tid + j * 256;
            if (TRANSB) {
                int n = v >> 3, c4 = v & 7;
                union { __bf16 h[4]; uint2 u; } t;
                t.h[0] = (__bf16)tb[j*4+0]; t.h[1] = (__bf16)tb[j*4+1];
                t.h[2] = (__bf16)tb[j*4+2]; t.h[3] = (__bf16)tb[j*4+3];
                *reinterpret_cast<uint2*>(&Bs[buf][n * LDB + c4 * 4]) = t.u;
            } else {
                int kk = v >> 4, c4 = v & 15;
                #pragma unroll
                for (int x = 0; x < 4; ++x)
                    Bs[buf][(c4 * 4 + x) * LDB + kk] = (__bf16)tb[j*4+x];
            }
        }
    };

    v8f zero = {};
    v8f acc[2][2];
    acc[0][0] = zero; acc[0][1] = zero; acc[1][0] = zero; acc[1][1] = zero;

    {
        float ta[16], tb[8];
        load_tiles(ta, tb, 0);
        stage(0, ta, tb);
    }
    __syncthreads();

    int cur = 0;
    for (int k0 = 0; k0 < K; k0 += BK) {
        const bool more = (k0 + BK) < K;
        float na[16], nb[8];
        if (more) load_tiles(na, nb, k0 + BK);   // overlap with WMMAs below

        Frag16 a[2], b[2];
        #pragma unroll
        for (int mf = 0; mf < 2; ++mf) {
            int row = wm * 32 + mf * 16 + l16;
            const __bf16* p = &As[cur][row * LDA + hl * 8];
            a[mf].q[0] = *reinterpret_cast<const uint4*>(p);
            a[mf].q[1] = *reinterpret_cast<const uint4*>(p + 16);
        }
        #pragma unroll
        for (int nf = 0; nf < 2; ++nf) {
            int col = wn * 32 + nf * 16 + l16;
            const __bf16* p = &Bs[cur][col * LDB + hl * 8];
            b[nf].q[0] = *reinterpret_cast<const uint4*>(p);
            b[nf].q[1] = *reinterpret_cast<const uint4*>(p + 16);
        }
        #pragma unroll
        for (int mf = 0; mf < 2; ++mf)
            #pragma unroll
            for (int nf = 0; nf < 2; ++nf)
                acc[mf][nf] = WMMA_BF16(a[mf].v, b[nf].v, acc[mf][nf]);

        if (more) {
            stage(1 - cur, na, nb);
            __syncthreads();
            cur ^= 1;
        }
    }

    // --- epilogue ---
    #pragma unroll
    for (int mf = 0; mf < 2; ++mf)
        #pragma unroll
        for (int nf = 0; nf < 2; ++nf)
            #pragma unroll
            for (int r = 0; r < 8; ++r) {
                int row = bm + wm * 32 + mf * 16 + r + hl * 8;
                int col = bn + wn * 32 + nf * 16 + l16;
                float v = acc[mf][nf][r];
                if (bias)  v += bias[col];
                if (resid) v += resid[(size_t)row * N + col];
                if (GELU)  v = gelu_exact(v);
                C[(size_t)row * N + col] = v;
            }
}

// ---------------------------------------------------------------------------
// Graph masked attention (per hop): 4 heads, hd=192, N=2048 nodes.
// One workgroup = one 32-query tile of one head (two 16-row WMMA tiles that
// share every K- and V-fragment: one fragment load feeds two WMMAs).
//  phase 1: scores (WMMA Q*K^T) -> 32x2048 fp32 slab in LDS (256KB), masked.
//  phase 2: rowwise softmax in LDS (unnormalized exp + row sums).
//  phase 3: P*V via WMMA (P read back from LDS in A-fragment layout).
// ---------------------------------------------------------------------------
__global__ __launch_bounds__(256) void graph_attn_kernel(
    const float* __restrict__ Q, const float* __restrict__ Kmat,
    const float* __restrict__ V, const unsigned char* __restrict__ adj,
    float* __restrict__ ctx)
{
    __shared__ float S[32][2048];
    __shared__ float red[32][9];
    __shared__ float rowmax[32];
    __shared__ float rowsum[32];

    const int qt   = blockIdx.x;        // 0..63
    const int head = blockIdx.y;        // 0..3
    const int tid  = threadIdx.x;
    const int w    = tid >> 5;
    const int lane = tid & 31;
    const int hl   = lane >> 4;
    const int l16  = lane & 15;
    const int q0   = qt * 32;
    const float rscale = 0.07216878364870323f;   // 1/sqrt(192)

    // preload Q fragments for both 16-row tiles (pre-scaled)
    Frag16 qa[2][6];
    #pragma unroll
    for (int t = 0; t < 2; ++t) {
        const float* qp = Q + (size_t)(q0 + t * 16 + l16) * 768 + head * 192;
        int kbase = hl * 8;
        #pragma unroll
        for (int s = 0; s < 6; ++s)
            #pragma unroll
            for (int e = 0; e < 8; ++e) {
                qa[t][s].e[e]     = (__bf16)(qp[s * 32 + kbase + e] * rscale);
                qa[t][s].e[8 + e] = (__bf16)(qp[s * 32 + kbase + 16 + e] * rscale);
            }
    }

    // phase 1: scores, masked, into LDS (each K fragment feeds 2 WMMAs)
    for (int kb = w; kb < 128; kb += 8) {
        int key0 = kb * 16;
        v8f acc0 = {}, acc1 = {};
        const float* kp = Kmat + (size_t)(key0 + l16) * 768 + head * 192 + hl * 8;
        #pragma unroll
        for (int s = 0; s < 6; ++s) {
            Frag16 bf;
            #pragma unroll
            for (int e = 0; e < 16; ++e) bf.e[e] = (__bf16)kp[s * 32 + (e < 8 ? e : e + 8)];
            acc0 = WMMA_BF16(qa[0][s].v, bf.v, acc0);
            acc1 = WMMA_BF16(qa[1][s].v, bf.v, acc1);
        }
        #pragma unroll
        for (int r = 0; r < 8; ++r) {
            int row0 = r + hl * 8;
            int row1 = 16 + row0;
            int key  = key0 + l16;
            bool m0 = adj[(size_t)(q0 + row0) * 2048 + key] != 0;
            bool m1 = adj[(size_t)(q0 + row1) * 2048 + key] != 0;
            S[row0][key] = m0 ? acc0[r] : -1e9f;
            S[row1][key] = m1 ? acc1[r] : -1e9f;
        }
    }
    __syncthreads();

    // phase 2: rowwise softmax (store unnormalized exp; keep row sums)
    {
        int r = tid >> 3, c = tid & 7;          // 32 rows x 8 lanes/row
        float pm = -3.0e38f;
        for (int key = c; key < 2048; key += 8) pm = fmaxf(pm, S[r][key]);
        red[r][c] = pm;
        __syncthreads();
        if (c == 0) {
            float m = red[r][0];
            #pragma unroll
            for (int i = 1; i < 8; ++i) m = fmaxf(m, red[r][i]);
            rowmax[r] = m;
        }
        __syncthreads();
        float m = rowmax[r];
        float ps = 0.0f;
        for (int key = c; key < 2048; key += 8) {
            float p = expf(S[r][key] - m);
            S[r][key] = p;
            ps += p;
        }
        red[r][c] = ps;
        __syncthreads();
        if (c == 0) {
            float s = 0.0f;
            #pragma unroll
            for (int i = 0; i < 8; ++i) s += red[r][i];
            rowsum[r] = s;
        }
        __syncthreads();
    }

    // phase 3: ctx = (P @ V) / rowsum ; each V fragment feeds 2 WMMAs
    for (int dt = w; dt < 12; dt += 8) {
        v8f acc0 = {}, acc1 = {};
        const float* vp = V + head * 192 + dt * 16 + l16;
        for (int kb = 0; kb < 64; ++kb) {
            Frag16 bf, a0, a1;
            int kB = kb * 32 + hl * 8;
            #pragma unroll
            for (int e = 0; e < 16; ++e)
                bf.e[e] = (__bf16)vp[(size_t)(kB + (e < 8 ? e : e + 8)) * 768];
            int kA = kb * 32 + hl * 8;
            #pragma unroll
            for (int e = 0; e < 8; ++e) {
                a0.e[e]     = (__bf16)S[l16][kA + e];
                a0.e[8 + e] = (__bf16)S[l16][kA + 16 + e];
                a1.e[e]     = (__bf16)S[16 + l16][kA + e];
                a1.e[8 + e] = (__bf16)S[16 + l16][kA + 16 + e];
            }
            acc0 = WMMA_BF16(a0.v, bf.v, acc0);
            acc1 = WMMA_BF16(a1.v, bf.v, acc1);
        }
        #pragma unroll
        for (int r = 0; r < 8; ++r) {
            int row0 = r + hl * 8;
            int row1 = 16 + row0;
            ctx[(size_t)(q0 + row0) * 768 + head * 192 + dt * 16 + l16] =
                acc0[r] / rowsum[row0];
            ctx[(size_t)(q0 + row1) * 768 + head * 192 + dt * 16 + l16] =
                acc1[r] / rowsum[row1];
        }
    }
}

// ---------------------------------------------------------------------------
// Knowledge attention: B=4, S=1024, NH=8, hd=96, 64 keys. One wave per
// (16-query tile, batch*head). Scores + PV both via WMMA.
// ---------------------------------------------------------------------------
__global__ __launch_bounds__(32) void kattn_kernel(
    const float* __restrict__ Q, const float* __restrict__ Kmat,
    const float* __restrict__ V, float* __restrict__ ctx)
{
    __shared__ float S[16][64];
    __shared__ float rowinv[16];

    const int qt  = blockIdx.x;          // 0..63
    const int bh  = blockIdx.y;          // 0..31
    const int b   = bh >> 3;
    const int h   = bh & 7;
    const int lane = threadIdx.x;
    const int hl   = lane >> 4;
    const int l16  = lane & 15;
    const int q0   = qt * 16;
    const float rscale = 0.10206207261596575f;   // 1/sqrt(96)

    Frag16 qa[3];
    {
        const float* qp = Q + (size_t)(b * 1024 + q0 + l16) * 768 + h * 96;
        int kbase = hl * 8;
        #pragma unroll
        for (int s = 0; s < 3; ++s)
            #pragma unroll
            for (int e = 0; e < 8; ++e) {
                qa[s].e[e]     = (__bf16)(qp[s * 32 + kbase + e] * rscale);
                qa[s].e[8 + e] = (__bf16)(qp[s * 32 + kbase + 16 + e] * rscale);
            }
    }

    #pragma unroll
    for (int kb = 0; kb < 4; ++kb) {
        int key0 = kb * 16;
        v8f acc = {};
        const float* kp = Kmat + (size_t)(b * 64 + key0 + l16) * 768 + h * 96 + hl * 8;
        #pragma unroll
        for (int s = 0; s < 3; ++s) {
            Frag16 bf;
            #pragma unroll
            for (int e = 0; e < 16; ++e) bf.e[e] = (__bf16)kp[s * 32 + (e < 8 ? e : e + 8)];
            acc = WMMA_BF16(qa[s].v, bf.v, acc);
        }
        #pragma unroll
        for (int r = 0; r < 8; ++r) S[r + hl * 8][key0 + l16] = acc[r];
    }
    __syncthreads();

    if (lane < 16) {
        float m = -3.0e38f;
        for (int k = 0; k < 64; ++k) m = fmaxf(m, S[lane][k]);
        float sum = 0.0f;
        for (int k = 0; k < 64; ++k) {
            float p = expf(S[lane][k] - m);
            S[lane][k] = p;
            sum += p;
        }
        rowinv[lane] = 1.0f / sum;
    }
    __syncthreads();

    #pragma unroll
    for (int dt = 0; dt < 6; ++dt) {
        v8f acc = {};
        const float* vp = V + (size_t)(b * 64) * 768 + h * 96 + dt * 16 + l16;
        #pragma unroll
        for (int kb = 0; kb < 2; ++kb) {
            Frag16 af, bf;
            int kA = kb * 32 + hl * 8;
            #pragma unroll
            for (int e = 0; e < 8; ++e) {
                af.e[e]     = (__bf16)S[l16][kA + e];
                af.e[8 + e] = (__bf16)S[l16][kA + 16 + e];
            }
            int kB = kb * 32 + hl * 8;
            #pragma unroll
            for (int e = 0; e < 16; ++e)
                bf.e[e] = (__bf16)vp[(size_t)(kB + (e < 8 ? e : e + 8)) * 768];
            acc = WMMA_BF16(af.v, bf.v, acc);
        }
        #pragma unroll
        for (int r = 0; r < 8; ++r) {
            int row = r + hl * 8;
            ctx[(size_t)(b * 1024 + q0 + row) * 768 + h * 96 + dt * 16 + l16] =
                acc[r] * rowinv[row];
        }
    }
}

// ---------------------------------------------------------------------------
// Rowwise softmax (in place), one block per row.
// ---------------------------------------------------------------------------
__global__ __launch_bounds__(256) void softmax_rows_kernel(float* __restrict__ X, int cols)
{
    __shared__ float red[256];
    const int row = blockIdx.x;
    const int tid = threadIdx.x;
    float* x = X + (size_t)row * cols;
    float m = -3.0e38f;
    for (int i = tid; i < cols; i += 256) m = fmaxf(m, x[i]);
    red[tid] = m; __syncthreads();
    for (int s = 128; s > 0; s >>= 1) { if (tid < s) red[tid] = fmaxf(red[tid], red[tid + s]); __syncthreads(); }
    m = red[0]; __syncthreads();
    float sum = 0.0f;
    for (int i = tid; i < cols; i += 256) { float p = expf(x[i] - m); x[i] = p; sum += p; }
    red[tid] = sum; __syncthreads();
    for (int s = 128; s > 0; s >>= 1) { if (tid < s) red[tid] += red[tid + s]; __syncthreads(); }
    float inv = 1.0f / red[0];
    __syncthreads();
    for (int i = tid; i < cols; i += 256) x[i] *= inv;
}

// ---------------------------------------------------------------------------
// LayerNorm over rows of length 768 (+ optional exact GELU afterwards).
// ---------------------------------------------------------------------------
template <bool GELU>
__global__ __launch_bounds__(256) void ln_kernel(
    const float* __restrict__ X, const float* __restrict__ g,
    const float* __restrict__ beta, float* __restrict__ out)
{
    __shared__ float red[256];
    const int row = blockIdx.x;
    const int tid = threadIdx.x;
    const float* x = X + (size_t)row * 768;
    float v[3];
    #pragma unroll
    for (int i = 0; i < 3; ++i) v[i] = x[tid + i * 256];
    float s = v[0] + v[1] + v[2];
    red[tid] = s; __syncthreads();
    for (int st = 128; st > 0; st >>= 1) { if (tid < st) red[tid] += red[tid + st]; __syncthreads(); }
    float mu = red[0] * (1.0f / 768.0f);
    __syncthreads();
    float sq = 0.0f;
    #pragma unroll
    for (int i = 0; i < 3; ++i) { float d = v[i] - mu; sq += d * d; }
    red[tid] = sq; __syncthreads();
    for (int st = 128; st > 0; st >>= 1) { if (tid < st) red[tid] += red[tid + st]; __syncthreads(); }
    float inv = rsqrtf(red[0] * (1.0f / 768.0f) + 1e-12f);
    #pragma unroll
    for (int i = 0; i < 3; ++i) {
        int col = tid + i * 256;
        float y = (v[i] - mu) * inv * g[col] + beta[col];
        if (GELU) y = gelu_exact(y);
        out[(size_t)row * 768 + col] = y;
    }
}

// ---------------------------------------------------------------------------
// Gather rows: dst[r,:] = src[ids[r],:]  (768 cols)
// ---------------------------------------------------------------------------
__global__ __launch_bounds__(256) void gather_kernel(
    const float* __restrict__ src, const int* __restrict__ ids, float* __restrict__ dst)
{
    const int row = blockIdx.x;
    const int id = ids[row];
    for (int i = threadIdx.x; i < 768; i += 256)
        dst[(size_t)row * 768 + i] = src[(size_t)id * 768 + i];
}

// ---------------------------------------------------------------------------
// Host orchestration
// ---------------------------------------------------------------------------
static inline void launch_gemm(const float* A, const float* B, const float* bias,
                               const float* resid, float* C, int M, int N, int K,
                               bool transb, bool gelu, hipStream_t s)
{
    dim3 grid(N / 64, M / 128), block(256);
    if (transb) {
        if (gelu) gemm_kernel<true, true ><<<grid, block, 0, s>>>(A, B, bias, resid, C, M, N, K);
        else      gemm_kernel<true, false><<<grid, block, 0, s>>>(A, B, bias, resid, C, M, N, K);
    } else {
        if (gelu) gemm_kernel<false, true ><<<grid, block, 0, s>>>(A, B, bias, resid, C, M, N, K);
        else      gemm_kernel<false, false><<<grid, block, 0, s>>>(A, B, bias, resid, C, M, N, K);
    }
}

extern "C" void kernel_launch(void* const* d_in, const int* in_sizes, int n_in,
                              void* d_out, int out_size, void* d_ws, size_t ws_size,
                              hipStream_t stream)
{
    (void)in_sizes; (void)n_in; (void)out_size; (void)ws_size;
    const int H = 768, N = 2048, HOPS = 3;
    const size_t HH = (size_t)H * H;

    const float* hidden    = (const float*)d_in[0];
    const float* etable    = (const float*)d_in[1];
    const float* gat_wq    = (const float*)d_in[2];
    const float* gat_wk    = (const float*)d_in[3];
    const float* gat_wv    = (const float*)d_in[4];
    const float* gat_wo    = (const float*)d_in[5];
    const float* gat_bq    = (const float*)d_in[6];
    const float* gat_bk    = (const float*)d_in[7];
    const float* gat_bv    = (const float*)d_in[8];
    const float* gat_bo    = (const float*)d_in[9];
    const float* gt_w      = (const float*)d_in[10];
    const float* gt_b      = (const float*)d_in[11];
    const float* gt_g      = (const float*)d_in[12];
    const float* gt_beta   = (const float*)d_in[13];
    const float* gp_w      = (const float*)d_in[14];
    const float* gp_b      = (const float*)d_in[15];
    const float* ka_wq     = (const float*)d_in[16];
    const float* ka_wk     = (const float*)d_in[17];
    const float* ka_wv     = (const float*)d_in[18];
    const float* ka_wo     = (const float*)d_in[19];
    const float* ka_bq     = (const float*)d_in[20];
    const float* ka_bk     = (const float*)d_in[21];
    const float* ka_bv     = (const float*)d_in[22];
    const float* ka_bo     = (const float*)d_in[23];
    const float* mem       = (const float*)d_in[24];
    const float* mem_wq    = (const float*)d_in[25];
    const float* mem_wm    = (const float*)d_in[26];
    const float* mem_wo    = (const float*)d_in[27];
    const float* mem_bq    = (const float*)d_in[28];
    const float* mem_bm    = (const float*)d_in[29];
    const float* mem_bo    = (const float*)d_in[30];
    const float* ff_w1     = (const float*)d_in[31];
    const float* ff_b1     = (const float*)d_in[32];
    const float* ff_w2     = (const float*)d_in[33];
    const float* ff_b2     = (const float*)d_in[34];
    const float* ff_g      = (const float*)d_in[35];
    const float* ff_beta   = (const float*)d_in[36];
    const float* ln1_g     = (const float*)d_in[37];
    const float* ln1_b     = (const float*)d_in[38];
    const float* ln2_g     = (const float*)d_in[39];
    const float* ln2_b     = (const float*)d_in[40];
    const int*   ids       = (const int*)d_in[41];
    const unsigned char* adj = (const unsigned char*)d_in[42];

    float* W = (float*)d_ws;
    constexpr size_t SZ_SMALL = 2048ull * 768;   // 1,572,864 floats
    constexpr size_t SZ_BIG   = 4096ull * 768;   // 3,145,728 floats
    float* NODE = W;
    float* QB   = W + 1 * SZ_SMALL;
    float* KB   = W + 2 * SZ_SMALL;
    float* VB   = W + 3 * SZ_SMALL;
    float* CTXB = W + 4 * SZ_SMALL;
    float* TB   = W + 5 * SZ_SMALL;
    float* R1   = W + 6 * SZ_SMALL;              // 3,145,728 floats
    float* R2   = R1 + SZ_BIG;
    float* R3   = R2 + SZ_BIG;
    float* R4   = R3 + SZ_BIG;                   // peak = 22,020,096 floats (~88 MB)
    // aliases (lifetimes verified disjoint):
    float* MQ  = NODE;   // 4096x768 spans NODE+QB
    float* MK  = KB;     // 1024x768
    float* SC  = VB;     // 4096x1024 spans VB+CTXB+TB
    float* FFA = NODE;   // 4096x2048 spans NODE..TB

    // ---- Graph reasoning hops ----
    const float* node = etable;
    for (int h = 0; h < HOPS; ++h) {
        launch_gemm(node, gat_wq + h * HH, gat_bq + h * H, nullptr, QB, N, H, H, false, false, stream);
        launch_gemm(node, gat_wk + h * HH, gat_bk + h * H, nullptr, KB, N, H, H, false, false, stream);
        launch_gemm(node, gat_wv + h * HH, gat_bv + h * H, nullptr, VB, N, H, H, false, false, stream);
        graph_attn_kernel<<<dim3(64, 4), 256, 0, stream>>>(QB, KB, VB, adj, CTXB);
        launch_gemm(CTXB, gat_wo + h * HH, gat_bo + h * H, node, TB, N, H, H, false, false, stream);
        launch_gemm(TB, gt_w + h * HH, gt_b + h * H, nullptr, QB, N, H, H, false, false, stream);
        ln_kernel<true><<<N, 256, 0, stream>>>(QB, gt_g + h * H, gt_beta + h * H, NODE);
        node = NODE;
    }
    // graph_out = node @ gp_w + gp_b  -> QB
    launch_gemm(node, gp_w, gp_b, nullptr, QB, N, H, H, false, false, stream);
    // knowledge_states gather -> KB (256 x 768)
    gather_kernel<<<256, 256, 0, stream>>>(QB, ids, KB);

    // ---- Knowledge attention ----
    launch_gemm(hidden, ka_wq, ka_bq, nullptr, R1, 4096, H, H, false, false, stream);  // KQ
    launch_gemm(KB, ka_wk, ka_bk, nullptr, VB, 256, H, H, false, false, stream);       // KK
    launch_gemm(KB, ka_wv, ka_bv, nullptr, CTXB, 256, H, H, false, false, stream);     // KV
    kattn_kernel<<<dim3(64, 32), 32, 0, stream>>>(R1, VB, CTXB, R2);                    // KCTX
    launch_gemm(R2, ka_wo, ka_bo, hidden, R3, 4096, H, H, false, false, stream);       // hidden + attn_out
    ln_kernel<false><<<4096, 256, 0, stream>>>(R3, ln1_g, ln1_b, R4);                  // h

    // ---- Memory bank ----
    launch_gemm(R4, mem_wq, mem_bq, nullptr, MQ, 4096, H, H, false, false, stream);
    launch_gemm(mem, mem_wm, mem_bm, nullptr, MK, 1024, H, H, false, false, stream);
    launch_gemm(MQ, MK, nullptr, nullptr, SC, 4096, 1024, H, true, false, stream);     // mq @ mk^T
    softmax_rows_kernel<<<4096, 256, 0, stream>>>(SC, 1024);
    launch_gemm(SC, mem, nullptr, nullptr, R1, 4096, H, 1024, false, false, stream);   // retrieved
    launch_gemm(R1, mem_wo, mem_bo, R4, R2, 4096, H, H, false, false, stream);         // h + mem_out
    ln_kernel<false><<<4096, 256, 0, stream>>>(R2, ln2_g, ln2_b, R3);                  // h2

    // ---- Feed forward ----
    launch_gemm(R3, ff_w1, ff_b1, nullptr, FFA, 4096, 2048, H, false, true, stream);   // gelu(h2@W1+b1)
    launch_gemm(FFA, ff_w2, ff_b2, R3, R1, 4096, H, 2048, false, false, stream);       // ff + h2
    ln_kernel<false><<<4096, 256, 0, stream>>>(R1, ff_g, ff_beta, (float*)d_out);
}
